// LongFormer_53919019434587
// MI455X (gfx1250) — matrix operand order repaired and checked by reading, not compiled, not run on hardware
//
#include <hip/hip_runtime.h>
#include <hip/hip_bf16.h>
#include <cstdint>

typedef __bf16 bf16;
typedef __attribute__((ext_vector_type(16))) __bf16 v16bf;
typedef __attribute__((ext_vector_type(8)))  float  v8f;
typedef unsigned int uint32;

// Fixed problem sizes
constexpr int Bc = 2, Sc = 4096, Dc = 768, Hc = 12, DHc = 64, NBc = 64, Fc = 3072, NSc = 8;
constexpr float SCALEc = 0.125f; // 1/sqrt(64)

// ---------------------------------------------------------------------------
// WMMA fragment helpers (CDNA5 16x16x32 bf16 layouts, wave32)
// A 16x32: lane half=l>>4, m=l&15; VGPR v holds K pair at (v>>2)*16 + half*8 + (v&3)*2
// B 32x16: lane half=l>>4, n=l&15; VGPR v holds K pair at (v>>2)*16 + half*8 + (v&3)*2  (element (k,n))
// C/D 16x16 f32: VGPR r -> row m = r + 8*half, col n = lane&15
// ---------------------------------------------------------------------------
__device__ inline v16bf frag_a(const bf16* base, int stride, int lane) {
    int half = lane >> 4, m = lane & 15;
    const bf16* row = base + m * stride;
    v16bf a;
#pragma unroll
    for (int v = 0; v < 8; ++v) {
        int kb = ((v >> 2) << 4) + (half << 3) + ((v & 3) << 1);
        a[2 * v]     = row[kb];
        a[2 * v + 1] = row[kb + 1];
    }
    return a;
}

// base points at element (k=0, n=0); element (k,n) lives at base[n*stride + k]
__device__ inline v16bf frag_b(const bf16* base, int stride, int lane) {
    int half = lane >> 4, n = lane & 15;
    const bf16* row = base + n * stride;
    v16bf b;
#pragma unroll
    for (int v = 0; v < 8; ++v) {
        int kb = ((v >> 2) << 4) + (half << 3) + ((v & 3) << 1);
        b[2 * v]     = row[kb];
        b[2 * v + 1] = row[kb + 1];
    }
    return b;
}

__device__ inline v8f wmma_bf16(v16bf a, v16bf b, v8f c) {
    return __builtin_amdgcn_wmma_f32_16x16x32_bf16(false, a, false, b, (short)0, c, false, false);
}

__device__ inline float block_reduce_sum(float v, float* red, int tid) {
    red[tid] = v;
    __syncthreads();
    for (int st = 128; st > 0; st >>= 1) {
        if (tid < st) red[tid] += red[tid + st];
        __syncthreads();
    }
    float r = red[0];
    __syncthreads();
    return r;
}

// ---------------------------------------------------------------------------
// One-time weight transpose + fp32->bf16: src [K][N] row-major -> dst [N][K]
// Block = 256 threads, 32x32 tile, both sides coalesced via LDS.
// ---------------------------------------------------------------------------
__global__ __launch_bounds__(256) void transpose_bf16_kernel(
    const float* __restrict__ src, bf16* __restrict__ dst, int K, int N)
{
    __shared__ bf16 t[32][33];
    int n0 = blockIdx.x * 32, k0 = blockIdx.y * 32;
    int tx = threadIdx.x & 31, ty = threadIdx.x >> 5;   // ty 0..7
#pragma unroll
    for (int i = 0; i < 4; ++i) {
        int k = ty + i * 8;
        t[tx][k] = (bf16)src[(size_t)(k0 + k) * N + n0 + tx];
    }
    __syncthreads();
#pragma unroll
    for (int i = 0; i < 4; ++i) {
        int n = ty + i * 8;
        dst[(size_t)(n0 + n) * K + k0 + tx] = t[n][tx];
    }
}

// ---------------------------------------------------------------------------
// Embedding + LayerNorm: one block per token, 256 threads, D=768 (3 per thread)
// ---------------------------------------------------------------------------
__global__ __launch_bounds__(256) void embed_ln_kernel(
    const int* __restrict__ ids, const int* __restrict__ tts,
    const float* __restrict__ we, const float* __restrict__ pe, const float* __restrict__ te,
    const float* __restrict__ g, const float* __restrict__ bta,
    float* __restrict__ hF, bf16* __restrict__ hB)
{
    __shared__ float red[256];
    int tok = blockIdx.x, tid = threadIdx.x;
    int s = tok & (Sc - 1);
    int id = ids[tok], tt = tts[tok];
    float v[3]; float sum = 0.f;
#pragma unroll
    for (int i = 0; i < 3; ++i) {
        int d = tid + i * 256;
        float x = we[(size_t)id * Dc + d] + pe[(size_t)s * Dc + d] + te[(size_t)tt * Dc + d];
        v[i] = x; sum += x;
    }
    float mean = block_reduce_sum(sum, red, tid) * (1.f / Dc);
    float s2 = 0.f;
#pragma unroll
    for (int i = 0; i < 3; ++i) { float d0 = v[i] - mean; s2 += d0 * d0; }
    float var = block_reduce_sum(s2, red, tid) * (1.f / Dc);
    float rstd = rsqrtf(var + 1e-12f);
#pragma unroll
    for (int i = 0; i < 3; ++i) {
        int d = tid + i * 256;
        float o = (v[i] - mean) * rstd * g[d] + bta[d];
        hF[(size_t)tok * Dc + d] = o;
        hB[(size_t)tok * Dc + d] = (bf16)o;
    }
}

// ---------------------------------------------------------------------------
// Residual + LayerNorm (x may alias hF; per-token block, reads precede writes)
// ---------------------------------------------------------------------------
__global__ __launch_bounds__(256) void add_ln_kernel(
    const float* x, const float* y,
    const float* __restrict__ g, const float* __restrict__ bta,
    float* hF, bf16* __restrict__ hB)
{
    __shared__ float red[256];
    int tok = blockIdx.x, tid = threadIdx.x;
    const float* xr = x + (size_t)tok * Dc;
    const float* yr = y + (size_t)tok * Dc;
    float v[3]; float sum = 0.f;
#pragma unroll
    for (int i = 0; i < 3; ++i) {
        int d = tid + i * 256;
        float t = xr[d] + yr[d];
        v[i] = t; sum += t;
    }
    float mean = block_reduce_sum(sum, red, tid) * (1.f / Dc);
    float s2 = 0.f;
#pragma unroll
    for (int i = 0; i < 3; ++i) { float d0 = v[i] - mean; s2 += d0 * d0; }
    float var = block_reduce_sum(s2, red, tid) * (1.f / Dc);
    float rstd = rsqrtf(var + 1e-12f);
#pragma unroll
    for (int i = 0; i < 3; ++i) {
        int d = tid + i * 256;
        float o = (v[i] - mean) * rstd * g[d] + bta[d];
        hF[(size_t)tok * Dc + d] = o;
        hB[(size_t)tok * Dc + d] = (bf16)o;
    }
}

// ---------------------------------------------------------------------------
// WMMA GEMM: C[M,N] = A[M,K](bf16) @ Wt[N,K](bf16, pre-transposed) + bias
// 128 threads = 4 waves. Workgroup tile 128x64, wave tile 32x64 (2 A-frags,
// 4 B-frags, 8 WMMAs per K-step; B fragment reused across 2 WMMAs).
// mode 0: linear out (outF f32 and/or outB bf16, optional GELU)
// mode 1: fused QKV scatter: n<768 -> Q [B,H,S,DH]; n<1536 -> K (same, +off);
//         else -> Vt [B,H,NB,DH,BLK] (+2*off). outB = base of contiguous QKV.
// ---------------------------------------------------------------------------
__global__ __launch_bounds__(128) void gemm_bf16_kernel(
    const bf16* __restrict__ A, const bf16* __restrict__ Wt, const float* __restrict__ bias,
    int M, int N, int K,
    float* __restrict__ outF, bf16* __restrict__ outB, int mode, int gelu)
{
    __shared__ bf16 lA[128 * 34];
    __shared__ bf16 lB[64 * 34];
    int tid = threadIdx.x;
    int lane = tid & 31, wave = tid >> 5;
    int bm = blockIdx.x * 128, bn = blockIdx.y * 64;

    v8f acc[2][4];
#pragma unroll
    for (int s = 0; s < 2; ++s)
#pragma unroll
        for (int t = 0; t < 4; ++t)
#pragma unroll
            for (int r = 0; r < 8; ++r) acc[s][t][r] = 0.f;

    for (int k0 = 0; k0 < K; k0 += 32) {
        // A tile 128x32 bf16 (16 x 4B per thread, coalesced)
        const bf16* ga = A + (size_t)bm * K + k0;
#pragma unroll
        for (int i = 0; i < 16; ++i) {
            int p = i * 128 + tid;               // pair index
            int r = p >> 4, c = (p & 15) << 1;
            *(uint32*)(&lA[r * 34 + c]) = *(const uint32*)(ga + (size_t)r * K + c);
        }
        // B tile: 64 rows (n) x 32 (k) from Wt[N][K]; pairs along K contiguous
        const bf16* gw = Wt + (size_t)bn * K + k0;
#pragma unroll
        for (int i = 0; i < 8; ++i) {
            int u = i * 128 + tid;               // pair index
            int n = u >> 4, c = (u & 15) << 1;
            *(uint32*)(&lB[n * 34 + c]) = *(const uint32*)(gw + (size_t)n * K + c);
        }
        __syncthreads();

        v16bf a0 = frag_a(&lA[(wave * 32) * 34], 34, lane);
        v16bf a1 = frag_a(&lA[(wave * 32 + 16) * 34], 34, lane);
#pragma unroll
        for (int t = 0; t < 4; ++t) {
            v16bf wf = frag_b(&lB[(t * 16) * 34], 34, lane);
            acc[0][t] = wmma_bf16(a0, wf, acc[0][t]);
            acc[1][t] = wmma_bf16(a1, wf, acc[1][t]);
        }
        __syncthreads();
    }

    const size_t qkvOff = (size_t)Bc * Sc * Dc;   // elements per Q/K/V tensor
    int half = lane >> 4, nl = lane & 15;
#pragma unroll
    for (int t = 0; t < 4; ++t) {
        int n = bn + t * 16 + nl;
        float bv = bias ? bias[n] : 0.f;
#pragma unroll
        for (int s = 0; s < 2; ++s) {
#pragma unroll
            for (int r = 0; r < 8; ++r) {
                int m = bm + wave * 32 + s * 16 + r + half * 8;
                float v = acc[s][t][r] + bv;
                if (gelu) {
                    float x3 = v * v * v;
                    v = 0.5f * v * (1.f + tanhf(0.79788456f * (v + 0.044715f * x3)));
                }
                if (mode == 0) {
                    if (outF) outF[(size_t)m * N + n] = v;
                    if (outB) outB[(size_t)m * N + n] = (bf16)v;
                } else {
                    int tq = n / 768;            // 0=Q, 1=K, 2=V
                    int e = n - tq * 768;
                    int b = m >> 12, sm = m & (Sc - 1);
                    int hh = e >> 6, dh = e & 63;
                    size_t base = (size_t)tq * qkvOff;
                    if (tq < 2) {
                        outB[base + (((size_t)(b * Hc + hh) * Sc + sm) << 6) + dh] = (bf16)v;
                    } else {
                        int nb = sm >> 6, qq = sm & 63;
                        outB[base + ((((size_t)(b * Hc + hh) * NBc + nb) * 64 + dh) << 6) + qq] = (bf16)v;
                    }
                }
            }
        }
    }
}

// ---------------------------------------------------------------------------
// Block-sparse flash attention. Grid (NB, H, B), 128 threads = 4 waves.
// Wave w handles query rows [w*16, w*16+16) of the 64-row query block.
// Q/K: [B,H,S,DH] bf16 row-major; V: [B,H,NB,DH,BLK] bf16 (transposed).
// Query blocks 0 and NB-1 attend densely to all NB key blocks.
// ---------------------------------------------------------------------------
__device__ inline int kv_block_id(bool dense, int nblk, int j, const int* randb) {
    if (dense) return j;
    if (j < 3) { int kb = nblk - 1 + j; return kb < 0 ? 0 : (kb > NBc - 1 ? NBc - 1 : kb); }
    if (j == 3) return 0;
    if (j == 4) return NBc - 1;
    return randb[nblk * 3 + (j - 5)];
}

__global__ __launch_bounds__(128) void attn_kernel(
    const bf16* __restrict__ Q, const bf16* __restrict__ Kg, const bf16* __restrict__ Vg,
    const float* __restrict__ amask, const int* __restrict__ randb,
    bf16* __restrict__ outB)
{
    __shared__ bf16 lQ[64 * 66];
    __shared__ bf16 lK[64 * 66];
    __shared__ bf16 lV[64 * 66];
    __shared__ bf16 lP[4][16 * 66];

    int nblk = blockIdx.x, h = blockIdx.y, b = blockIdx.z;
    int tid = threadIdx.x, lane = tid & 31, wave = tid >> 5;
    int half = lane >> 4, nl = lane & 15;
    const size_t headOff = (size_t)(b * Hc + h) * (size_t)Sc * DHc;

    // Stage the query block: lQ[q*66 + dh]
    {
        const bf16* gq = Q + headOff + (size_t)nblk * 64 * 64;
#pragma unroll
        for (int i = 0; i < 16; ++i) {
            int flat = (i * 128 + tid) * 2;
            int r = flat >> 6, c = flat & 63;
            *(uint32*)(&lQ[r * 66 + c]) = *(const uint32*)(gq + r * 64 + c);
        }
    }

    float m_i[8], l_i[8];
    v8f accO[4];
#pragma unroll
    for (int r = 0; r < 8; ++r) { m_i[r] = -1e30f; l_i[r] = 0.f; }
#pragma unroll
    for (int t = 0; t < 4; ++t)
#pragma unroll
        for (int r = 0; r < 8; ++r) accO[t][r] = 0.f;

    bool dense = (nblk == 0) || (nblk == NBc - 1);
    int nkv = dense ? NBc : 8;

    for (int j = 0; j < nkv; ++j) {
        int kb = kv_block_id(dense, nblk, j, randb);

        __syncthreads();
        // Stage K block (lK[key*66+dh]) and Vt block (lV[dh*66+key])
        const bf16* gk = Kg + headOff + (size_t)kb * 64 * 64;
        const bf16* gv = Vg + headOff + (size_t)kb * 64 * 64;
#pragma unroll
        for (int i = 0; i < 16; ++i) {
            int flat = (i * 128 + tid) * 2;
            int r = flat >> 6, c = flat & 63;
            *(uint32*)(&lK[r * 66 + c]) = *(const uint32*)(gk + r * 64 + c);
            *(uint32*)(&lV[r * 66 + c]) = *(const uint32*)(gv + r * 64 + c);
        }
        __syncthreads();

        // Prefetch next KV block while computing this one (global_prefetch_b8)
        if (j + 1 < nkv) {
            int kbn = kv_block_id(dense, nblk, j + 1, randb);
            const bf16* pk = Kg + headOff + (size_t)kbn * 64 * 64 + tid * 32;
            const bf16* pv = Vg + headOff + (size_t)kbn * 64 * 64 + tid * 32;
            __builtin_prefetch(pk, 0, 1);
            __builtin_prefetch(pv, 0, 1);
        }

        // Scores: 16x64 tile per wave. A = Q rows, B = K (already pair-contiguous)
        v8f accS[4];
#pragma unroll
        for (int t = 0; t < 4; ++t)
#pragma unroll
            for (int r = 0; r < 8; ++r) accS[t][r] = 0.f;

        const bf16* qBase = &lQ[(wave * 16) * 66];
#pragma unroll
        for (int ks = 0; ks < 64; ks += 32) {
            v16bf aq = frag_a(qBase + ks, 66, lane);
#pragma unroll
            for (int t = 0; t < 4; ++t) {
                v16bf bk = frag_b(&lK[(t * 16) * 66] + ks, 66, lane);
                accS[t] = wmma_bf16(aq, bk, accS[t]);
            }
        }

        // Per-column mask penalty
        float mvals[4];
#pragma unroll
        for (int t = 0; t < 4; ++t)
            mvals[t] = (amask[b * Sc + kb * 64 + t * 16 + nl] - 1.f) * 1e9f;

        // Online softmax per owned row (rows live across 16 lanes of one half)
#pragma unroll
        for (int r = 0; r < 8; ++r) {
            float sv[4]; float mx = -1e30f;
#pragma unroll
            for (int t = 0; t < 4; ++t) {
                sv[t] = accS[t][r] * SCALEc + mvals[t];
                mx = fmaxf(mx, sv[t]);
            }
            for (int off = 1; off < 16; off <<= 1) mx = fmaxf(mx, __shfl_xor(mx, off, 32));
            float mnew = fmaxf(m_i[r], mx);
            float corr = __expf(m_i[r] - mnew);
            float rs = 0.f;
            float pv[4];
#pragma unroll
            for (int t = 0; t < 4; ++t) { pv[t] = __expf(sv[t] - mnew); rs += pv[t]; }
            for (int off = 1; off < 16; off <<= 1) rs += __shfl_xor(rs, off, 32);
            l_i[r] = l_i[r] * corr + rs;
            m_i[r] = mnew;
#pragma unroll
            for (int t = 0; t < 4; ++t) accO[t][r] *= corr;
            // write P (bf16) to this wave's LDS tile: lP[row*66 + key]
            int qrow = r + (half << 3);
            bf16* prow = &lP[wave][qrow * 66];
#pragma unroll
            for (int t = 0; t < 4; ++t) prow[t * 16 + nl] = (bf16)pv[t];
        }

        // Same-wave LDS RAW: make P visible before fragment reload
        asm volatile("s_wait_dscnt 0" ::: "memory");

        // O += P @ V. A = P rows (pairs along key), B = Vt (pairs along key)
        const bf16* pBase = &lP[wave][0];
#pragma unroll
        for (int ks = 0; ks < 64; ks += 32) {
            v16bf ap = frag_a(pBase + ks, 66, lane);
#pragma unroll
            for (int t = 0; t < 4; ++t) {
                v16bf bv = frag_b(&lV[(t * 16) * 66] + ks, 66, lane);
                accO[t] = wmma_bf16(ap, bv, accO[t]);
            }
        }
    }

    // Normalize and write to [B,S,D] bf16 (column = h*64+dh)
#pragma unroll
    for (int t = 0; t < 4; ++t) {
        int dh = t * 16 + nl;
#pragma unroll
        for (int r = 0; r < 8; ++r) {
            int row = wave * 16 + r + (half << 3);
            int s = nblk * 64 + row;
            float o = accO[t][r] / l_i[r];
            outB[((size_t)(b * Sc + s)) * Dc + h * 64 + dh] = (bf16)o;
        }
    }
}

// ---------------------------------------------------------------------------
// Span mean pooling + classification head. One block per (b, span).
// ---------------------------------------------------------------------------
__global__ __launch_bounds__(256) void pool_head_kernel(
    const float* __restrict__ hF, const int* __restrict__ spans,
    const float* __restrict__ head_w, const float* __restrict__ head_b,
    float* __restrict__ out)
{
    __shared__ float r0[256], r1[256];
    int idx = blockIdx.x;          // b*NS + n
    int tid = threadIdx.x;
    int b = idx >> 3, n = idx & 7;
    int st = spans[(b * NSc + n) * 2 + 0];
    int en = spans[(b * NSc + n) * 2 + 1];
    float acc[3] = {0.f, 0.f, 0.f};
    for (int s = st; s < en; ++s) {
        const float* row = hF + ((size_t)(b * Sc + s)) * Dc;
#pragma unroll
        for (int i = 0; i < 3; ++i) acc[i] += row[tid + i * 256];
    }
    float inv = 1.f / (float)(en - st);
    float p0 = 0.f, p1 = 0.f;
#pragma unroll
    for (int i = 0; i < 3; ++i) {
        int d = tid + i * 256;
        float e = acc[i] * inv;
        p0 += e * head_w[d * 2 + 0];
        p1 += e * head_w[d * 2 + 1];
    }
    r0[tid] = p0; r1[tid] = p1;
    __syncthreads();
    for (int s2 = 128; s2 > 0; s2 >>= 1) {
        if (tid < s2) { r0[tid] += r0[tid + s2]; r1[tid] += r1[tid + s2]; }
        __syncthreads();
    }
    if (tid == 0) {
        out[idx * 2 + 0] = r0[0] + head_b[0];
        out[idx * 2 + 1] = r1[0] + head_b[1];
    }
}

// ---------------------------------------------------------------------------
extern "C" void kernel_launch(void* const* d_in, const int* in_sizes, int n_in,
                              void* d_out, int out_size, void* d_ws, size_t ws_size,
                              hipStream_t stream) {
    (void)in_sizes; (void)n_in; (void)out_size; (void)ws_size;
    const int*   input_ids = (const int*)  d_in[0];
    const float* amask     = (const float*)d_in[1];
    const int*   ttids     = (const int*)  d_in[2];
    const int*   spans     = (const int*)  d_in[3];
    const int*   randb     = (const int*)  d_in[4];
    const float* word_emb  = (const float*)d_in[5];
    const float* pos_emb   = (const float*)d_in[6];
    const float* type_emb  = (const float*)d_in[7];
    const float* emb_g     = (const float*)d_in[8];
    const float* emb_b     = (const float*)d_in[9];
    const float* qkv_w     = (const float*)d_in[10];
    const float* qkv_b     = (const float*)d_in[11];
    const float* aout_w    = (const float*)d_in[12];
    const float* aout_b    = (const float*)d_in[13];
    const float* ln1_g     = (const float*)d_in[14];
    const float* ln1_b     = (const float*)d_in[15];
    const float* ff1_w     = (const float*)d_in[16];
    const float* ff1_b     = (const float*)d_in[17];
    const float* ff2_w     = (const float*)d_in[18];
    const float* ff2_b     = (const float*)d_in[19];
    const float* ln2_g     = (const float*)d_in[20];
    const float* ln2_b     = (const float*)d_in[21];
    const float* head_w    = (const float*)d_in[22];
    const float* head_b    = (const float*)d_in[23];

    const size_t MT = (size_t)Bc * Sc;      // 8192 tokens
    uintptr_t cur = (uintptr_t)d_ws;
    auto alloc = [&](size_t bytes) -> void* {
        uintptr_t p = (cur + 255) & ~(uintptr_t)255;
        cur = p + bytes;
        return (void*)p;
    };
    float* hF    = (float*)alloc(MT * Dc * sizeof(float));
    float* yF    = (float*)alloc(MT * Dc * sizeof(float));
    bf16*  hB    = (bf16*) alloc(MT * Dc * sizeof(bf16));
    bf16*  qkvB  = (bf16*) alloc(3 * MT * Dc * sizeof(bf16));  // Q | K | Vt contiguous
    bf16*  attnB = (bf16*) alloc(MT * Dc * sizeof(bf16));
    bf16*  ffB   = (bf16*) alloc(MT * Fc * sizeof(bf16));
    // Pre-transposed bf16 weights: Wt[N][K]
    bf16*  WtQKV = (bf16*) alloc((size_t)2 * 3 * Dc * Dc * sizeof(bf16)); // [L][2304][768]
    bf16*  WtA   = (bf16*) alloc((size_t)2 * Dc * Dc * sizeof(bf16));     // [L][768][768]
    bf16*  WtF1  = (bf16*) alloc((size_t)2 * Fc * Dc * sizeof(bf16));     // [L][3072][768]
    bf16*  WtF2  = (bf16*) alloc((size_t)2 * Dc * Fc * sizeof(bf16));     // [L][768][3072]

    bf16* qB = qkvB;
    bf16* kB = qkvB + MT * Dc;
    bf16* vB = qkvB + 2 * MT * Dc;

    // One-time weight conversion (cheap: ~28 MB written once per call)
    for (int l = 0; l < 2; ++l) {
        for (int t = 0; t < 3; ++t) {
            transpose_bf16_kernel<<<dim3(Dc / 32, Dc / 32), 256, 0, stream>>>(
                qkv_w + ((size_t)(l * 3 + t)) * Dc * Dc,
                WtQKV + (size_t)l * 3 * Dc * Dc + (size_t)t * Dc * Dc, Dc, Dc);
        }
        transpose_bf16_kernel<<<dim3(Dc / 32, Dc / 32), 256, 0, stream>>>(
            aout_w + (size_t)l * Dc * Dc, WtA + (size_t)l * Dc * Dc, Dc, Dc);
        transpose_bf16_kernel<<<dim3(Fc / 32, Dc / 32), 256, 0, stream>>>(
            ff1_w + (size_t)l * Dc * Fc, WtF1 + (size_t)l * Fc * Dc, Dc, Fc);
        transpose_bf16_kernel<<<dim3(Dc / 32, Fc / 32), 256, 0, stream>>>(
            ff2_w + (size_t)l * Fc * Dc, WtF2 + (size_t)l * Dc * Fc, Fc, Dc);
    }

    embed_ln_kernel<<<(int)MT, 256, 0, stream>>>(input_ids, ttids, word_emb, pos_emb,
                                                 type_emb, emb_g, emb_b, hF, hB);

    dim3 gQKV((int)(MT / 128), (3 * Dc) / 64);  // 64 x 36  (N = 2304)
    dim3 gD((int)(MT / 128), Dc / 64);          // 64 x 12  (N = 768)
    dim3 gF((int)(MT / 128), Fc / 64);          // 64 x 48  (N = 3072)
    dim3 gA(NBc, Hc, Bc);                       // (64, 12, 2)

    for (int l = 0; l < 2; ++l) {
        // Fused Q/K/V projection with layout scatter
        gemm_bf16_kernel<<<gQKV, 128, 0, stream>>>(
            hB, WtQKV + (size_t)l * 3 * Dc * Dc, qkv_b + (size_t)l * 3 * Dc,
            (int)MT, 3 * Dc, Dc, nullptr, qkvB, 1, 0);

        attn_kernel<<<gA, 128, 0, stream>>>(qB, kB, vB, amask, randb, attnB);

        gemm_bf16_kernel<<<gD, 128, 0, stream>>>(
            attnB, WtA + (size_t)l * Dc * Dc, aout_b + (size_t)l * Dc,
            (int)MT, Dc, Dc, yF, nullptr, 0, 0);
        add_ln_kernel<<<(int)MT, 256, 0, stream>>>(hF, yF, ln1_g + l * Dc, ln1_b + l * Dc, hF, hB);

        gemm_bf16_kernel<<<gF, 128, 0, stream>>>(
            hB, WtF1 + (size_t)l * Fc * Dc, ff1_b + (size_t)l * Fc,
            (int)MT, Fc, Dc, nullptr, ffB, 0, 1);
        gemm_bf16_kernel<<<gD, 128, 0, stream>>>(
            ffB, WtF2 + (size_t)l * Dc * Fc, ff2_b + (size_t)l * Dc,
            (int)MT, Dc, Fc, yF, nullptr, 0, 0);
        add_ln_kernel<<<(int)MT, 256, 0, stream>>>(hF, yF, ln2_g + l * Dc, ln2_b + l * Dc, hF, hB);
    }

    pool_head_kernel<<<Bc * NSc, 256, 0, stream>>>(hF, spans, head_w, head_b, (float*)d_out);
}
